// Model_70918499991669
// MI455X (gfx1250) — matrix operand order repaired and checked by reading, not compiled, not run on hardware
//
#include <hip/hip_runtime.h>

// Problem constants (from reference): B,T,H,K,V = 2,4096,16,128,128
#define TB  2
#define TT  4096
#define TH  16
#define TK  128
#define TV  128
#define CC  16            // chunk length (time)
#define NCH (TT / CC)     // 256 chunks
#define VBW 16            // V-columns per wave
#define WPB 8             // waves per block (cover full V = 8*16)

typedef __attribute__((ext_vector_type(16))) __bf16 v16bf;
typedef __attribute__((ext_vector_type(8)))  float  v8f;

union BF16x16 { unsigned int u[8]; v16bf v; };

__device__ __forceinline__ unsigned short f2bf_bits(float f) {
  __bf16 h = (__bf16)f;                       // native bf16 convert on CDNA5
  return __builtin_bit_cast(unsigned short, h);
}

#if __has_builtin(__builtin_amdgcn_cvt_pk_bf16_f32)
__device__ __forceinline__ unsigned int pack2(float lo, float hi) {
  auto p = __builtin_amdgcn_cvt_pk_bf16_f32(lo, hi);   // v_cvt_pk_bf16_f32
  static_assert(sizeof(p) == 4, "pk bf16 must be 32-bit");
  return __builtin_bit_cast(unsigned int, p);
}
#else
__device__ __forceinline__ unsigned int pack2(float lo, float hi) {
  return (unsigned int)f2bf_bits(lo) | ((unsigned int)f2bf_bits(hi) << 16);
}
#endif

__device__ __forceinline__ v8f wmma_bf16(v16bf a, v16bf b, v8f c) {
  // D = A(16x32 bf16) * B(32x16 bf16) + C(16x16 f32)
  return __builtin_amdgcn_wmma_f32_16x16x32_bf16(
      false, a, false, b, (short)0, c, false, false);
}

// A-operand tile (16x32, bf16) per ISA 7.12.2: lane m=lo, half=hi.
// regs 0-3 -> K = 2j + 8*hi ; regs 4-7 -> K = 16 + 2(j-4) + 8*hi
__device__ __forceinline__ v16bf ldA_tile(const unsigned short* arr, int stride,
                                          int jt, int lo, int hi) {
  BF16x16 r;
#pragma unroll
  for (int j = 0; j < 8; ++j) {
    const int kl = (j < 4) ? (2 * j + 8 * hi) : (16 + 2 * (j - 4) + 8 * hi);
    r.u[j] = *(const unsigned int*)(arr + lo * stride + 32 * jt + kl);
  }
  return r.v;
}

// B-operand tile (32x16, bf16): lane n=lo (column), regs hold K pairs:
// regs 0-3 -> K = 2j + 8*hi ; regs 4-7 -> K = 16 + 2(j-4) + 8*hi
__device__ __forceinline__ v16bf ldB_tile(const unsigned short* arr, int stride,
                                          int jt, int lo, int hi) {
  BF16x16 r;
#pragma unroll
  for (int j = 0; j < 8; ++j) {
    r.u[j] = *(const unsigned int*)(arr + lo * stride + 32 * jt + ((j < 4) ? (2 * j + 8 * hi) : (16 + 2 * (j - 4) + 8 * hi)));
  }
  return r.v;
}

__global__ __launch_bounds__(256, 1)
void kda_chunk_kernel(const float* __restrict__ q, const float* __restrict__ k,
                      const float* __restrict__ v, const float* __restrict__ g,
                      const float* __restrict__ beta, float* __restrict__ out) {
  // Shared staging, DOUBLE-BUFFERED (one __syncthreads per chunk suffices:
  // reaching staging of chunk c+2, which reuses buffer c&1, requires passing
  // the barrier of chunk c+1, which requires all waves done reading c&1).
  __shared__ unsigned short s_kx [2][CC][TK];   // kappa  = k*exp(+cumg)       bf16
  __shared__ unsigned short s_qx [2][CC][TK];   // qtilde = scale*q*exp(+cumg) bf16
  __shared__ unsigned short s_eta[2][CC][TK];   // eta    = beta*k*exp(-cumg)  bf16
  __shared__ unsigned short s_etaT[2][TK][CC];  // eta, k-major                bf16
  __shared__ float          s_gam[2][TK];       // Gamma_C per key channel
  // Per-wave private regions (same-wave DS is in-order; no barrier needed).
  __shared__ unsigned short s_sb [WPB][VBW][TK]; // state B-operand image [v][k]
  __shared__ float          s_A  [WPB][CC][CC];  // kappa@eta^T (strict-lower)
  __shared__ float          s_Aq [WPB][CC][CC];  // qtilde@eta^T (incl-lower)
  __shared__ float          s_u  [WPB][CC][VBW]; // rhs / solved U (f32)
  __shared__ unsigned short s_ub [WPB][VBW][CC]; // U bf16, [v][s]

  const int tid  = threadIdx.x;
  const int w    = tid >> 5;            // wave id = V-block id
  const int lane = tid & 31;
  const int lo   = lane & 15;
  const int hi   = (lane >> 4) & 1;
  const int h    = blockIdx.x % TH;
  const int b    = blockIdx.x / TH;

  const float scale = 0.08838834764831845f;   // 128^-0.5

  // Staging (waves 0..3): lane owns one k-channel for the whole chunk, since
  // the cumulative-sum over t is sequential; 32 lanes x 4 waves = 128 chans.
  const int  kc   = 32 * w + lane;      // valid when w < 4
  const int  qbase = (b * TT * TH + h) * TK;
  const float* gp = g + qbase + kc;
  const float* kp = k + qbase + kc;
  const float* qp = q + qbase + kc;
  const float* bp = beta + b * TT * TH + h;

  // Compute-side pointers (per wave: 16 v-columns starting at w*16).
  const int vbase = (b * TT * TH + h) * TV + w * VBW;
  const float* vp = v + vbase + lo + 8 * hi * TH * TV;
  float*       op = out + vbase + lo + 8 * hi * TH * TV;

  // State slab: K=128 rows x V=16 cols as 8 WMMA C/D tiles (f32).
  // Tile ik covers k-rows [16ik,16ik+16); C-layout: reg r -> k=16ik+r+8hi, col=lo.
  v8f S[8];
#pragma unroll
  for (int ik = 0; ik < 8; ++ik)
    S[ik] = (v8f){0.f, 0.f, 0.f, 0.f, 0.f, 0.f, 0.f, 0.f};

  for (int ch = 0; ch < NCH; ++ch) {
    const int bsel = ch & 1;

    // ---- Pre-barrier, per-wave work: state -> bf16 image, preload v -------
#pragma unroll
    for (int ik = 0; ik < 8; ++ik) {
#pragma unroll
      for (int r = 0; r < 8; r += 2) {
        *(unsigned int*)&s_sb[w][lo][16 * ik + 8 * hi + r] =
            pack2(S[ik][r], S[ik][r + 1]);
      }
    }
    float vvv[8];
#pragma unroll
    for (int r = 0; r < 8; ++r) vvv[r] = vp[r * TH * TV];

    // ---- Staging (waves 0..3 only): cumsum + decayed operands ------------
    if (w < 4) {
      if (ch + 1 < NCH) {
        __builtin_prefetch(gp + CC * TH * TK, 0, 1);
        __builtin_prefetch(kp + CC * TH * TK, 0, 1);
        __builtin_prefetch(qp + CC * TH * TK, 0, 1);
      }
      float bet[CC];
#pragma unroll
      for (int t = 0; t < CC; ++t) bet[t] = bp[t * TH];   // scalar loads
      float run = 0.f, ep = 1.f;
#pragma unroll
      for (int t = 0; t < CC; ++t) {
        const float gv = gp[t * TH * TK];
        const float kv = kp[t * TH * TK];
        const float qv = qp[t * TH * TK];
        run += gv;                                    // cumulative log-decay
        ep = __expf(run);                             // Gamma_t (<= 1)
        const float em = __builtin_amdgcn_rcpf(ep);   // Gamma_t^-1 (bf16 range ok)
        const unsigned short eb = f2bf_bits(bet[t] * kv * em);
        s_kx [bsel][t][kc] = f2bf_bits(kv * ep);
        s_qx [bsel][t][kc] = f2bf_bits(scale * qv * ep);
        s_eta[bsel][t][kc] = eb;
        s_etaT[bsel][kc][t] = eb;
      }
      s_gam[bsel][kc] = ep;                           // Gamma_C
      gp += CC * TH * TK;  kp += CC * TH * TK;  qp += CC * TH * TK;
      bp += CC * TH;
    }

    __syncthreads();    // staging of buffer bsel complete

    // ---- Phase 4: chunk-level matmuls over K=128 (4 x 16x16x32) ----------
    v8f vhat = {0.f,0.f,0.f,0.f,0.f,0.f,0.f,0.f};
    v8f Am   = {0.f,0.f,0.f,0.f,0.f,0.f,0.f,0.f};
    v8f Aqm  = {0.f,0.f,0.f,0.f,0.f,0.f,0.f,0.f};
    v8f Oacc = {0.f,0.f,0.f,0.f,0.f,0.f,0.f,0.f};
#pragma unroll
    for (int jt = 0; jt < 4; ++jt) {
      const v16bf a_kx  = ldA_tile(&s_kx [bsel][0][0], TK, jt, lo, hi);
      const v16bf a_qx  = ldA_tile(&s_qx [bsel][0][0], TK, jt, lo, hi);
      const v16bf b_eta = ldB_tile(&s_eta[bsel][0][0], TK, jt, lo, hi);
      const v16bf b_sb  = ldB_tile(&s_sb [w][0][0],    TK, jt, lo, hi);
      Am   = wmma_bf16(a_kx, b_eta, Am);    // A   = kappa  @ eta^T
      vhat = wmma_bf16(a_kx, b_sb,  vhat);  // v^0 = kappa  @ S0
      Aqm  = wmma_bf16(a_qx, b_eta, Aqm);   // Aq  = qtilde @ eta^T
      Oacc = wmma_bf16(a_qx, b_sb,  Oacc);  // O  += qtilde @ S0
    }

    // Spill A/Aq and rhs = V - vhat (C-layout: reg r -> t=r+8hi, col=lo)
#pragma unroll
    for (int r = 0; r < 8; ++r) {
      const int t = r + 8 * hi;
      s_A [w][t][lo] = Am[r];
      s_Aq[w][t][lo] = Aqm[r];
      s_u [w][t][lo] = vvv[r] - vhat[r];
    }

    // ---- Phase 5: forward substitution U = (I + A_strict)^-1 rhs ---------
    if (hi == 0) {                      // 16 lanes, lane = v column
      float uv[CC];
#pragma unroll
      for (int t = 0; t < CC; ++t) {
        float x = s_u[w][t][lo];
#pragma unroll
        for (int s = 0; s < t; ++s) x -= s_A[w][t][s] * uv[s];
        uv[t] = x;
        s_ub[w][lo][t] = f2bf_bits(x);
      }
    }
    // (reconverged: EXEC all-1 again before any WMMA)

    // U as B operand (regs 0-3: K = s = 2j + 8*hi; regs 4-7: K=16..31 zero pad)
    BF16x16 bu;
#pragma unroll
    for (int j = 0; j < 8; ++j)
      bu.u[j] = (j >= 4) ? 0u : *(const unsigned int*)&s_ub[w][lo][2 * j + 8 * hi];

    // ---- Phase 6: intra-chunk output O += (mask .* Aq) @ U ---------------
    BF16x16 wt;
#pragma unroll
    for (int j = 0; j < 8; ++j) {
      if (j < 4) {
        const int s0 = 2 * j + 8 * hi;
        const float w0 = (s0     <= lo) ? s_Aq[w][lo][s0]     : 0.f;
        const float w1 = (s0 + 1 <= lo) ? s_Aq[w][lo][s0 + 1] : 0.f;
        wt.u[j] = pack2(w0, w1);
      } else {
        wt.u[j] = 0u;                   // K=16..31 zero padding
      }
    }
    Oacc = wmma_bf16(wt.v, bu.v, Oacc);

#pragma unroll
    for (int r = 0; r < 8; ++r)
      op[r * TH * TV] = Oacc[r];

    // ---- Phase 7: state update S = Diag(Gamma_C)*(S0 + eta^T @ U) --------
#pragma unroll
    for (int ik = 0; ik < 8; ++ik) {
      BF16x16 ae;                       // etaT tile [16k x 16s], K padded to 32
#pragma unroll
      for (int j = 0; j < 8; ++j) {
        ae.u[j] = (j < 4)
            ? *(const unsigned int*)&s_etaT[bsel][16 * ik + lo][2 * j + 8 * hi]
            : 0u;
      }
      S[ik] = wmma_bf16(ae.v, bu.v, S[ik]);
    }
#pragma unroll
    for (int ik = 0; ik < 8; ++ik) {
#pragma unroll
      for (int r = 0; r < 8; ++r)
        S[ik][r] *= s_gam[bsel][16 * ik + 8 * hi + r];
    }

    vp += CC * TH * TV;
    op += CC * TH * TV;
  }
}

extern "C" void kernel_launch(void* const* d_in, const int* in_sizes, int n_in,
                              void* d_out, int out_size, void* d_ws, size_t ws_size,
                              hipStream_t stream) {
  (void)in_sizes; (void)n_in; (void)d_ws; (void)ws_size; (void)out_size;
  const float* q    = (const float*)d_in[0];
  const float* k    = (const float*)d_in[1];
  const float* v    = (const float*)d_in[2];
  const float* g    = (const float*)d_in[3];
  const float* beta = (const float*)d_in[4];
  float* out        = (float*)d_out;

  dim3 grid(TB * TH);       // one 8-wave workgroup per (b,h)
  dim3 block(32 * WPB);     // 8 wave32s
  hipLaunchKernelGGL(kda_chunk_kernel, grid, block, 0, stream,
                     q, k, v, g, beta, out);
}
